// PVMLayer4_6511170421519
// MI455X (gfx1250) — compile-verified, matrix operand order checked by reading, not run on hardware
//
#include <hip/hip_runtime.h>
#include <hip/hip_bf16.h>
#include <cstdint>
#include <cstddef>

// ---------------- CDNA5 WMMA types ----------------
typedef __bf16 bf16_t;
typedef __attribute__((ext_vector_type(8)))  __bf16 v8bf;
typedef __attribute__((ext_vector_type(16))) __bf16 v16bf;
typedef __attribute__((ext_vector_type(8)))  float  v8f;

union V16 { v16bf v; v8bf h[2]; };

// ---------------- problem constants ----------------
constexpr int B_SZ   = 4;
constexpr int LSEQ   = 4096;           // 64*64
constexpr int BL     = B_SZ * LSEQ;    // 16384 tokens
constexpr int DSTATE = 16;
constexpr int XDLD   = 48;             // x_dbl padded stride (36 -> 48)

// workspace layout (element offsets)
constexpr size_t OFF_XN     = 0;                              // BL*256 f32
constexpr size_t OFF_CONCAT = OFF_XN     + (size_t)BL*256;    // BL*256 f32
constexpr size_t OFF_XZ     = OFF_CONCAT + (size_t)BL*256;    // BL*256 f32
constexpr size_t OFF_UF     = OFF_XZ     + (size_t)BL*256;    // BL*128 f32
constexpr size_t OFF_DT     = OFF_UF     + (size_t)BL*128;    // BL*128 f32
constexpr size_t OFF_XDBL   = OFF_DT     + (size_t)BL*128;    // BL*48  f32
constexpr size_t OFF_YF     = OFF_XDBL   + (size_t)BL*XDLD;   // BL*128 f32
constexpr size_t F32_TOTAL  = OFF_YF     + (size_t)BL*128;

constexpr size_t OFF_XNB  = 0;                              // BL*256 bf16
constexpr size_t OFF_UB   = OFF_XNB  + (size_t)BL*256;      // BL*128 bf16
constexpr size_t OFF_YB   = OFF_UB   + (size_t)BL*128;      // BL*128 bf16
constexpr size_t OFF_LN2B = OFF_YB   + (size_t)BL*128;      // BL*256 bf16
constexpr size_t OFF_INWB = OFF_LN2B + (size_t)BL*256;      // 256*64
constexpr size_t OFF_XPWB = OFF_INWB + 256*64;              // 48*128 (zero-padded)
constexpr size_t OFF_OPWB = OFF_XPWB + 48*128;              // 64*128
constexpr size_t OFF_PWB  = OFF_OPWB + 64*128;              // 256*256

__device__ __forceinline__ float silu_f(float x) { return x / (1.0f + __expf(-x)); }
__device__ __forceinline__ float softplus_f(float x) {
    return (x > 20.0f) ? x : log1pf(__expf(x));
}

// -------- weights f32 -> bf16 (x_proj zero-padded to 48 rows) --------
__global__ void k_convert(const float* __restrict__ inw, const float* __restrict__ xpw,
                          const float* __restrict__ opw, const float* __restrict__ pw,
                          bf16_t* __restrict__ inwb, bf16_t* __restrict__ xpwb,
                          bf16_t* __restrict__ opwb, bf16_t* __restrict__ pwb) {
    int idx = blockIdx.x * blockDim.x + threadIdx.x;
    const int N1 = 256 * 64, N2 = 48 * 128, N3 = 64 * 128, N4 = 256 * 256;
    if (idx < N1) {
        inwb[idx] = (bf16_t)inw[idx];
    } else if (idx < N1 + N2) {
        int i = idx - N1, j = i >> 7, k = i & 127;
        xpwb[i] = (bf16_t)((j < 36) ? xpw[j * 128 + k] : 0.0f);
    } else if (idx < N1 + N2 + N3) {
        int i = idx - N1 - N2;
        opwb[i] = (bf16_t)opw[i];
    } else if (idx < N1 + N2 + N3 + N4) {
        int i = idx - N1 - N2 - N3;
        pwb[i] = (bf16_t)pw[i];
    }
}

// -------- LN1: (B,256,L) channel-major -> token-major LN via LDS transpose --------
__global__ void k_ln1(const float* __restrict__ x, const float* __restrict__ w,
                      const float* __restrict__ bn, float* __restrict__ xn,
                      bf16_t* __restrict__ xnb) {
    __shared__ float tile[32 * 257];
    const int b  = blockIdx.x >> 7;
    const int l0 = (blockIdx.x & 127) * 32;
    const int li = threadIdx.x & 31;
    const int c0 = threadIdx.x >> 5;
    // coalesced gather of a 32(token) x 256(ch) tile
    #pragma unroll
    for (int i = 0; i < 32; ++i) {
        int c = i * 8 + c0;
        tile[li * 257 + c] = x[((size_t)b * 256 + c) * LSEQ + l0 + li];
    }
    __syncthreads();
    const int wave = threadIdx.x >> 5;
    const int lane = threadIdx.x & 31;
    for (int j = 0; j < 4; ++j) {
        int ll = wave * 4 + j;
        float vals[8], s = 0.f, ss = 0.f;
        #pragma unroll
        for (int i = 0; i < 8; ++i) {
            float v = tile[ll * 257 + i * 32 + lane];
            vals[i] = v; s += v; ss += v * v;
        }
        #pragma unroll
        for (int m = 16; m >= 1; m >>= 1) {
            s  += __shfl_xor(s, m);
            ss += __shfl_xor(ss, m);
        }
        float mu = s * (1.0f / 256.0f);
        float var = ss * (1.0f / 256.0f) - mu * mu;
        float rs = rsqrtf(var + 1e-5f);
        size_t t = (size_t)b * LSEQ + l0 + ll;
        #pragma unroll
        for (int i = 0; i < 8; ++i) {
            int c = i * 32 + lane;
            float o = (vals[i] - mu) * rs * w[c] + bn[c];
            xn[t * 256 + c]  = o;
            xnb[t * 256 + c] = (bf16_t)o;
        }
    }
}

// -------- bf16 WMMA GEMM, 16x(16*NT) tile per wave:
//          C[M,N] = A[M,K] * W[N,K]^T (+bias)(+skip) --------
template<int NT>
__global__ void k_gemm_bf16(const bf16_t* __restrict__ A, int lda, int aoff,
                            const bf16_t* __restrict__ W, int K,
                            float* __restrict__ C, int ldc, int coff, int nGroups,
                            const float* __restrict__ bias,
                            const float* __restrict__ skip, int lskip, int soff,
                            const float* __restrict__ sscale) {
    const int wave   = blockIdx.x * (blockDim.x >> 5) + (threadIdx.x >> 5);
    const int lane   = threadIdx.x & 31;
    const int ngrp   = wave % nGroups;
    const int mtile  = wave / nGroups;
    const int ntile0 = ngrp * NT;
    const int half   = lane >> 4;
    const int r      = lane & 15;
    const bf16_t* Arow = A + (size_t)(mtile * 16 + r) * lda + aoff;
    const bf16_t* Wrow = W + (size_t)(ntile0 * 16 + r) * K;
    v8f acc[NT];
    #pragma unroll
    for (int j = 0; j < NT; ++j) acc[j] = (v8f){};
    for (int kb = 0; kb < K; kb += 32) {
        int base = kb + half * 8;
        V16 av;
        av.h[0] = *(const v8bf*)(Arow + base);
        av.h[1] = *(const v8bf*)(Arow + base + 16);
        #pragma unroll
        for (int j = 0; j < NT; ++j) {
            V16 bv;
            bv.h[0] = *(const v8bf*)(Wrow + (size_t)j * 16 * K + base);
            bv.h[1] = *(const v8bf*)(Wrow + (size_t)j * 16 * K + base + 16);
            acc[j] = __builtin_amdgcn_wmma_f32_16x16x32_bf16(false, av.v, false, bv.v,
                                                             (short)0, acc[j], false, false);
        }
    }
    const float sc = skip ? sscale[0] : 0.0f;
    #pragma unroll
    for (int j = 0; j < NT; ++j) {
        const int col = (ntile0 + j) * 16 + r;
        const float bval = bias ? bias[col] : 0.0f;
        #pragma unroll
        for (int v = 0; v < 8; ++v) {
            int row = mtile * 16 + half * 8 + v;
            float val = acc[j][v] + bval;
            if (skip) val += sc * skip[(size_t)row * lskip + soff + col];
            C[(size_t)row * ldc + coff + col] = val;
        }
    }
}

// -------- causal depthwise conv(4) + SiLU --------
__global__ void k_conv(const float* __restrict__ xz, const float* __restrict__ cw,
                       const float* __restrict__ cb, float* __restrict__ u,
                       bf16_t* __restrict__ ub) {
    int idx = blockIdx.x * blockDim.x + threadIdx.x;   // BL*128
    int d = idx & 127, t = idx >> 7;
    int b = t >> 12, l = t & (LSEQ - 1);
    float acc = cb[d];
    #pragma unroll
    for (int k = 0; k < 4; ++k) {
        int ll = l - 3 + k;
        if (ll >= 0) acc += xz[((size_t)(b * LSEQ + ll)) * 256 + d] * cw[d * 4 + k];
    }
    acc = silu_f(acc);
    u[(size_t)t * 128 + d]  = acc;
    ub[(size_t)t * 128 + d] = (bf16_t)acc;
}

// -------- dt = softplus(dt_r @ dt_proj_w^T + b) --------
__global__ void k_dt(const float* __restrict__ xdbl, const float* __restrict__ dpw,
                     const float* __restrict__ dpb, float* __restrict__ dt) {
    int idx = blockIdx.x * blockDim.x + threadIdx.x;   // BL*128
    int d = idx & 127, t = idx >> 7;
    const float* xd = &xdbl[(size_t)t * XDLD];
    float v = dpb[d];
    #pragma unroll
    for (int r = 0; r < 4; ++r) v += xd[r] * dpw[d * 4 + r];
    dt[(size_t)t * 128 + d] = softplus_f(v);
}

// -------- selective scan: lane = (batch, d_inner), 16 states in registers --------
__global__ void k_scan(const float* __restrict__ xz, const float* __restrict__ u,
                       const float* __restrict__ dt, const float* __restrict__ xdbl,
                       const float* __restrict__ A_log, const float* __restrict__ Dp,
                       float* __restrict__ y, bf16_t* __restrict__ yb) {
    const int b = blockIdx.x >> 2;
    const int d = (blockIdx.x & 3) * 32 + threadIdx.x;
    float A[DSTATE], h[DSTATE];
    #pragma unroll
    for (int n = 0; n < DSTATE; ++n) {
        A[n] = -__expf(A_log[d * DSTATE + n]);
        h[n] = 0.0f;
    }
    const float Dd = Dp[d];
    for (int l = 0; l < LSEQ; ++l) {
        size_t t = (size_t)b * LSEQ + l;
        __builtin_prefetch(&xdbl[(t + 8) * XDLD], 0, 0);
        float dtv = dt[t * 128 + d];
        float uv  = u[t * 128 + d];
        float zv  = xz[t * 256 + 128 + d];
        const float* bc = &xdbl[t * XDLD + 4];  // B[0..15], C[16..31]
        float dtu = dtv * uv, yv = 0.0f;
        #pragma unroll
        for (int n = 0; n < DSTATE; ++n) {
            float dA = __expf(dtv * A[n]);
            h[n] = h[n] * dA + dtu * bc[n];
            yv += h[n] * bc[DSTATE + n];
        }
        yv = (yv + uv * Dd) * silu_f(zv);
        y[t * 128 + d]  = yv;
        yb[t * 128 + d] = (bf16_t)yv;
    }
}

// -------- LN2 over token-major concat -> bf16 --------
__global__ void k_ln2(const float* __restrict__ src, const float* __restrict__ w,
                      const float* __restrict__ bn, bf16_t* __restrict__ dst) {
    const size_t t = (size_t)blockIdx.x * 8 + (threadIdx.x >> 5);
    const int lane = threadIdx.x & 31;
    float vals[8], s = 0.f, ss = 0.f;
    #pragma unroll
    for (int i = 0; i < 8; ++i) {
        float v = src[t * 256 + i * 32 + lane];
        vals[i] = v; s += v; ss += v * v;
    }
    #pragma unroll
    for (int m = 16; m >= 1; m >>= 1) {
        s  += __shfl_xor(s, m);
        ss += __shfl_xor(ss, m);
    }
    float mu = s * (1.0f / 256.0f);
    float var = ss * (1.0f / 256.0f) - mu * mu;
    float rs = rsqrtf(var + 1e-5f);
    #pragma unroll
    for (int i = 0; i < 8; ++i) {
        int c = i * 32 + lane;
        dst[t * 256 + c] = (bf16_t)((vals[i] - mu) * rs * w[c] + bn[c]);
    }
}

// -------- final 256x256 WMMA GEMM, 16x64 per wave,
//          LDS-transposed coalesced (B,C,L) store --------
__global__ void k_gemm_final(const bf16_t* __restrict__ A, const bf16_t* __restrict__ W,
                             const float* __restrict__ bias, float* __restrict__ out) {
    __shared__ float sm[8][16 * 17];
    const int wave   = blockIdx.x * (blockDim.x >> 5) + (threadIdx.x >> 5);
    const int wv     = threadIdx.x >> 5;
    const int lane   = threadIdx.x & 31;
    const int ngrp   = wave & 3;          // 4 groups of 4 n-tiles = 256 cols
    const int mtile  = wave >> 2;
    const int ntile0 = ngrp * 4;
    const int half   = lane >> 4;
    const int r      = lane & 15;
    const bf16_t* Arow = A + (size_t)(mtile * 16 + r) * 256;
    const bf16_t* Wrow = W + (size_t)(ntile0 * 16 + r) * 256;
    v8f acc[4];
    #pragma unroll
    for (int j = 0; j < 4; ++j) acc[j] = (v8f){};
    for (int kb = 0; kb < 256; kb += 32) {
        int base = kb + half * 8;
        V16 av;
        av.h[0] = *(const v8bf*)(Arow + base);
        av.h[1] = *(const v8bf*)(Arow + base + 16);
        #pragma unroll
        for (int j = 0; j < 4; ++j) {
            V16 bv;
            bv.h[0] = *(const v8bf*)(Wrow + (size_t)j * 16 * 256 + base);
            bv.h[1] = *(const v8bf*)(Wrow + (size_t)j * 16 * 256 + base + 16);
            acc[j] = __builtin_amdgcn_wmma_f32_16x16x32_bf16(false, av.v, false, bv.v,
                                                             (short)0, acc[j], false, false);
        }
    }
    // per-wave LDS transpose: 8 scattered b32 stores/lane -> 2 coalesced b128 stores/lane
    const int t0 = mtile * 16;            // 16 tokens, never crosses a batch boundary
    const int bb = t0 >> 12;
    const int l0 = t0 & (LSEQ - 1);
    const int o_local = lane >> 1;
    const int lh      = lane & 1;
    #pragma unroll
    for (int j = 0; j < 4; ++j) {
        #pragma unroll
        for (int v = 0; v < 8; ++v)
            sm[wv][(half * 8 + v) * 17 + r] = acc[j][v];
        // same-wave DS ops are in-order: no barrier needed
        const int o = (ntile0 + j) * 16 + o_local;
        const float bval = bias[o];
        float e[8];
        #pragma unroll
        for (int i = 0; i < 8; ++i)
            e[i] = sm[wv][(lh * 8 + i) * 17 + o_local] + bval;
        float* p = out + ((size_t)bb * 256 + o) * LSEQ + l0 + lh * 8;
        *(float4*)(p)     = make_float4(e[0], e[1], e[2], e[3]);
        *(float4*)(p + 4) = make_float4(e[4], e[5], e[6], e[7]);
    }
}

extern "C" void kernel_launch(void* const* d_in, const int* in_sizes, int n_in,
                              void* d_out, int out_size, void* d_ws, size_t ws_size,
                              hipStream_t stream) {
    const float* x        = (const float*)d_in[0];
    const float* norm_w   = (const float*)d_in[1];
    const float* norm_b   = (const float*)d_in[2];
    const float* in_proj  = (const float*)d_in[3];
    const float* conv_w   = (const float*)d_in[4];
    const float* conv_b   = (const float*)d_in[5];
    const float* x_proj   = (const float*)d_in[6];
    const float* dt_pw    = (const float*)d_in[7];
    const float* dt_pb    = (const float*)d_in[8];
    const float* A_log    = (const float*)d_in[9];
    const float* Dp       = (const float*)d_in[10];
    const float* out_proj = (const float*)d_in[11];
    const float* proj_w   = (const float*)d_in[12];
    const float* proj_b   = (const float*)d_in[13];
    const float* sscale   = (const float*)d_in[14];
    float* out = (float*)d_out;

    float* fw = (float*)d_ws;
    bf16_t* bw = (bf16_t*)(fw + F32_TOTAL);

    float*  XN     = fw + OFF_XN;
    float*  CONCAT = fw + OFF_CONCAT;
    float*  XZ     = fw + OFF_XZ;
    float*  UF     = fw + OFF_UF;
    float*  DT     = fw + OFF_DT;
    float*  XDBL   = fw + OFF_XDBL;
    float*  YF     = fw + OFF_YF;
    bf16_t* XNB  = bw + OFF_XNB;
    bf16_t* UB   = bw + OFF_UB;
    bf16_t* YB   = bw + OFF_YB;
    bf16_t* LN2B = bw + OFF_LN2B;
    bf16_t* INWB = bw + OFF_INWB;
    bf16_t* XPWB = bw + OFF_XPWB;
    bf16_t* OPWB = bw + OFF_OPWB;
    bf16_t* PWB  = bw + OFF_PWB;

    // weights -> bf16 (x_proj zero-padded to N=48)
    k_convert<<<(256*64 + 48*128 + 64*128 + 256*256 + 255) / 256, 256, 0, stream>>>(
        in_proj, x_proj, out_proj, proj_w, INWB, XPWB, OPWB, PWB);

    // LN1 with LDS transpose
    k_ln1<<<B_SZ * (LSEQ / 32), 256, 0, stream>>>(x, norm_w, norm_b, XN, XNB);

    for (int c = 0; c < 4; ++c) {
        // in_proj: (BL,64) x (256,64)^T -> XZ (BL,256); 16x64 per wave, 4 groups
        k_gemm_bf16<4><<<(BL / 16) * 4 / 8, 256, 0, stream>>>(
            XNB, 256, 64 * c, INWB, 64, XZ, 256, 0, 4,
            nullptr, nullptr, 0, 0, nullptr);
        // conv + silu
        k_conv<<<BL * 128 / 256, 256, 0, stream>>>(XZ, conv_w, conv_b, UF, UB);
        // x_proj: (BL,128) x (48,128)^T -> XDBL (BL,48); 16x48 per wave
        k_gemm_bf16<3><<<(BL / 16) * 1 / 8, 256, 0, stream>>>(
            UB, 128, 0, XPWB, 128, XDBL, XDLD, 0, 1,
            nullptr, nullptr, 0, 0, nullptr);
        // dt
        k_dt<<<BL * 128 / 256, 256, 0, stream>>>(XDBL, dt_pw, dt_pb, DT);
        // selective scan + gate
        k_scan<<<16, 32, 0, stream>>>(XZ, UF, DT, XDBL, A_log, Dp, YF, YB);
        // out_proj + skip into concat columns [64c, 64c+64); 16x64 per wave
        k_gemm_bf16<4><<<(BL / 16) * 1 / 8, 256, 0, stream>>>(
            YB, 128, 0, OPWB, 128, CONCAT, 256, 64 * c, 1,
            nullptr, XN, 256, 64 * c, sscale);
    }

    // LN2 + final projection with LDS-transposed coalesced store
    k_ln2<<<BL / 8, 256, 0, stream>>>(CONCAT, norm_w, norm_b, LN2B);
    k_gemm_final<<<(BL / 16) * 4 / 8, 256, 0, stream>>>(LN2B, PWB, proj_b, out);
}
